// TASTGCN_71270687310384
// MI455X (gfx1250) — matrix operand order repaired and checked by reading, not compiled, not run on hardware
//
#include <hip/hip_runtime.h>
#include <hip/hip_bf16.h>

#define ROWS 196608L   // B*N*T = 16*1024*12

typedef __attribute__((ext_vector_type(16))) __bf16 v16bf;
typedef __attribute__((ext_vector_type(8)))  __bf16 v8bf;
typedef __attribute__((ext_vector_type(8)))  float  v8f;

__device__ __forceinline__ float sigf(float x){ return 1.f/(1.f+__expf(-x)); }

// ---------------------------------------------------------------------------
// Generic batched WMMA GEMM:  out[b] = act(A[b] (MxK) * Bt[b]^T + bias)
// A rows contiguous (lda == K). Bt stored as Bt[n*K + k] (row per output
// column, contiguous K) -> fragments load as contiguous v8bf vectors.
// One wave computes a 32x32 macro tile = 2x2 WMMA tiles: each A/B fragment is
// reused 2x -> 2 b128 loads per v_wmma (vs 4 for a single-tile wave), halving
// L2 traffic on the K=1024 graph/fuse mixes, while keeping register pressure
// low enough (4 accum octets + 4 fragment octets) to avoid spills (the 4x4
// variant spilled accumulators around every WMMA).
// Requires M%32==0, N%32==0, K%32==0 (true for every call site here).
// ---------------------------------------------------------------------------
__global__ __launch_bounds__(256, 1)
void gemm_wmma_bf16(const __bf16* __restrict__ A, long strideAb,
                    const __bf16* __restrict__ Bt, long strideBb,
                    int M, int Nout, int K,
                    const float* __restrict__ bias, int act,
                    float* __restrict__ outF, long strideOFb,
                    __bf16* __restrict__ outB, long strideOBb)
{
    const int b       = blockIdx.y;
    const int wavesPB = blockDim.x >> 5;
    const int wid     = blockIdx.x * wavesPB + (threadIdx.x >> 5);
    const int ntiles  = Nout >> 5;          // 32-wide macro tiles
    const int mtiles  = M >> 5;
    if (wid >= ntiles * mtiles) return;
    const int tm   = (wid / ntiles) << 5;
    const int tn   = (wid % ntiles) << 5;
    const int lane = threadIdx.x & 31;
    const int hh   = lane >> 4;             // half-wave id
    const int lrow = lane & 15;

    // A fragment: lane holds row; half 0 covers K=k0+{0..7,16..23},
    // half 1 covers K=k0+{8..15,24..31}   (ISA 16-bit A 16x32 layout)
    const __bf16* ar0 = A  + (long)b*strideAb + ((long)(tm + lrow))*K + hh*8;
    // B fragment: lane holds column; half 0 K=k0+{0..7,16..23}, half 1 K=k0+{8..15,24..31}
    const __bf16* br0 = Bt + (long)b*strideBb + ((long)(tn + lrow))*K + hh*8;
    const long step16 = 16L * K;            // next 16-row/col block

    v8f c00 = (v8f){}, c01 = (v8f){}, c10 = (v8f){}, c11 = (v8f){};

    for (int k0 = 0; k0 < K; k0 += 32) {
        v8bf a00 = *(const v8bf*)(ar0 + k0);
        v8bf a01 = *(const v8bf*)(ar0 + k0 + 16);
        v8bf a10 = *(const v8bf*)(ar0 + step16 + k0);
        v8bf a11 = *(const v8bf*)(ar0 + step16 + k0 + 16);
        v8bf b00 = *(const v8bf*)(br0 + k0);
        v8bf b01 = *(const v8bf*)(br0 + k0 + 16);
        v8bf b10 = *(const v8bf*)(br0 + step16 + k0);
        v8bf b11 = *(const v8bf*)(br0 + step16 + k0 + 16);
        v16bf af0 = __builtin_shufflevector(a00, a01, 0,1,2,3,4,5,6,7,8,9,10,11,12,13,14,15);
        v16bf af1 = __builtin_shufflevector(a10, a11, 0,1,2,3,4,5,6,7,8,9,10,11,12,13,14,15);
        v16bf bf0 = __builtin_shufflevector(b00, b01, 0,1,2,3,4,5,6,7,8,9,10,11,12,13,14,15);
        v16bf bf1 = __builtin_shufflevector(b10, b11, 0,1,2,3,4,5,6,7,8,9,10,11,12,13,14,15);
        c00 = __builtin_amdgcn_wmma_f32_16x16x32_bf16(false, af0, false, bf0, (short)0, c00, false, false);
        c01 = __builtin_amdgcn_wmma_f32_16x16x32_bf16(false, af0, false, bf1, (short)0, c01, false, false);
        c10 = __builtin_amdgcn_wmma_f32_16x16x32_bf16(false, af1, false, bf0, (short)0, c10, false, false);
        c11 = __builtin_amdgcn_wmma_f32_16x16x32_bf16(false, af1, false, bf1, (short)0, c11, false, false);
    }

    const int col0 = tn + lrow;
    const int col1 = tn + 16 + lrow;
    const float badd0 = bias ? bias[col0] : 0.f;
    const float badd1 = bias ? bias[col1] : 0.f;
    #pragma unroll
    for (int jj = 0; jj < 8; ++jj) {
        const int row0 = tm + jj + hh*8;        // C/D: VGPR jj -> M = jj + 8*half
        const int row1 = tm + 16 + jj + hh*8;
        float v00 = c00[jj] + badd0, v01 = c01[jj] + badd1;
        float v10 = c10[jj] + badd0, v11 = c11[jj] + badd1;
        if (act == 1) { v00 = fmaxf(v00, 0.f); v01 = fmaxf(v01, 0.f);
                        v10 = fmaxf(v10, 0.f); v11 = fmaxf(v11, 0.f); }
        const long o00 = (long)row0 * Nout + col0, o01 = (long)row0 * Nout + col1;
        const long o10 = (long)row1 * Nout + col0, o11 = (long)row1 * Nout + col1;
        if (outF) {
            float* of = outF + (long)b*strideOFb;
            of[o00] = v00; of[o01] = v01; of[o10] = v10; of[o11] = v11;
        }
        if (outB) {
            __bf16* ob = outB + (long)b*strideOBb;
            ob[o00] = (__bf16)v00; ob[o01] = (__bf16)v01;
            ob[o10] = (__bf16)v10; ob[o11] = (__bf16)v11;
        }
    }
}

// ------------------------------ prep kernels -------------------------------
__global__ void f2bf_kernel(const float* __restrict__ s, __bf16* __restrict__ d, long n){
    long i = blockIdx.x*(long)blockDim.x + threadIdx.x;
    if (i < n) d[i] = (__bf16)s[i];
}
// gc_weight is (d,e) used as x@W -> need Wt[e*64+d]
__global__ void transpose64_kernel(const float* __restrict__ s, __bf16* __restrict__ d){
    int i = blockIdx.x*blockDim.x + threadIdx.x; if (i >= 4096) return;
    int r = i >> 6, cc = i & 63;
    d[cc*64 + r] = (__bf16)s[r*64 + cc];
}
// conv1x3 weights (O,64,1,3) -> Wt[o*192 + kt*64 + i]; o<Oa from wA else wB
__global__ void conv3_reorder_kernel(const float* __restrict__ wA, const float* __restrict__ wB,
                                     int Oa, int Ototal, __bf16* __restrict__ dst){
    int i = blockIdx.x*blockDim.x + threadIdx.x;
    if (i >= Ototal*192) return;
    int o = i / 192, kk = i % 192, kt = kk / 64, ii = kk % 64;
    const float* src = (o < Oa) ? (wA + (long)o*192) : (wB + (long)(o-Oa)*192);
    dst[(long)o*192 + kt*64 + ii] = (__bf16)src[ii*3 + kt];
}
__global__ void concat2_kernel(const float* __restrict__ a, const float* __restrict__ b,
                               int na, int ntot, float* __restrict__ dst){
    int i = blockIdx.x*blockDim.x + threadIdx.x; if (i >= ntot) return;
    dst[i] = (i < na) ? a[i] : b[i - na];
}

// A_comb[m,:] = softmax(relu(A1[m,:])) + softmax(relu(A2[m,:]))  (axis=1)
__global__ void softmax_add_kernel(const float* __restrict__ A1, const float* __restrict__ A2,
                                   __bf16* __restrict__ out){
    const int m = blockIdx.x, tid = threadIdx.x;          // 256 threads
    __shared__ float e1[1024], e2[1024], red[256];
    for (int i = tid; i < 1024; i += 256){
        e1[i] = fmaxf(A1[(long)m*1024 + i], 0.f);
        e2[i] = fmaxf(A2[(long)m*1024 + i], 0.f);
    }
    __syncthreads();
    float mx1 = -1e30f, mx2 = -1e30f;
    for (int i = tid; i < 1024; i += 256){ mx1 = fmaxf(mx1, e1[i]); mx2 = fmaxf(mx2, e2[i]); }
    red[tid] = mx1; __syncthreads();
    for (int s = 128; s > 0; s >>= 1){ if (tid < s) red[tid] = fmaxf(red[tid], red[tid+s]); __syncthreads(); }
    mx1 = red[0]; __syncthreads();
    red[tid] = mx2; __syncthreads();
    for (int s = 128; s > 0; s >>= 1){ if (tid < s) red[tid] = fmaxf(red[tid], red[tid+s]); __syncthreads(); }
    mx2 = red[0]; __syncthreads();
    float s1 = 0.f, s2 = 0.f;
    for (int i = tid; i < 1024; i += 256){
        float a = __expf(e1[i]-mx1); e1[i] = a; s1 += a;
        float c = __expf(e2[i]-mx2); e2[i] = c; s2 += c;
    }
    red[tid] = s1; __syncthreads();
    for (int s = 128; s > 0; s >>= 1){ if (tid < s) red[tid] += red[tid+s]; __syncthreads(); }
    s1 = red[0]; __syncthreads();
    red[tid] = s2; __syncthreads();
    for (int s = 128; s > 0; s >>= 1){ if (tid < s) red[tid] += red[tid+s]; __syncthreads(); }
    s2 = red[0]; __syncthreads();
    const float i1 = 1.f/s1, i2 = 1.f/s2;
    for (int i = tid; i < 1024; i += 256)
        out[(long)m*1024 + i] = (__bf16)(e1[i]*i1 + e2[i]*i2);
}

// m[n] = mean_b relu(flow[b,n,:] . cl_w + cl_b)
__global__ void node_mean_kernel(const float* __restrict__ flow, const float* __restrict__ clw,
                                 const float* __restrict__ clb, float* __restrict__ mvec){
    int n = blockIdx.x*blockDim.x + threadIdx.x; if (n >= 1024) return;
    float acc = 0.f;
    for (int b = 0; b < 16; ++b){
        const float* f = flow + ((long)b*1024 + n)*12;
        float s = clb[0];
        for (int t = 0; t < 12; ++t) s += f[t]*clw[t];
        acc += fmaxf(s, 0.f);
    }
    mvec[n] = acc * (1.f/16.f);
}
__global__ void fuse_mat_kernel(const float* __restrict__ mvec, __bf16* __restrict__ fuse){
    long i = blockIdx.x*(long)blockDim.x + threadIdx.x; if (i >= 1024L*1024) return;
    int mm = (int)(i >> 10), nn = (int)(i & 1023);
    fuse[i] = (__bf16)sigf(0.5f*(mvec[mm] + mvec[nn]));
}

// xin[row,64] = [flow | spatial | daily[idx] | weekly[idx]]
__global__ void embed_kernel(const float* __restrict__ flow, const int* __restrict__ day,
                             const int* __restrict__ week, const float* __restrict__ sp,
                             const float* __restrict__ de, const float* __restrict__ we,
                             __bf16* __restrict__ xin){
    long i = blockIdx.x*(long)blockDim.x + threadIdx.x; if (i >= ROWS*64) return;
    int d = (int)(i & 63); long row = i >> 6;
    long bn = row / 12; int n = (int)(bn & 1023);
    float v;
    if      (d == 0)  v = flow[row];
    else if (d <= 32) v = sp[(long)n*32 + (d-1)];
    else if (d <= 48) v = de[(long)day[row]*16 + (d-33)];
    else              v = we[(long)week[row]*15 + (d-49)];
    xin[i] = (__bf16)v;
}

// x0 = lhs * sigmoid(rhs); write transposed x0t[b][t*64+d][n]
__global__ void glu_combine_kernel(const float* __restrict__ g, __bf16* __restrict__ x0t){
    long i = blockIdx.x*(long)blockDim.x + threadIdx.x; if (i >= ROWS*64) return;
    int d = (int)(i & 63); long row = i >> 6;
    float v = g[row*128 + d] * sigf(g[row*128 + 64 + d]);
    int t = (int)(row % 12); long bn = row / 12;
    int n = (int)(bn & 1023); int b = (int)(bn >> 10);
    x0t[((long)b*768 + t*64 + d)*1024 + n] = (__bf16)v;
}

// gather time-shifted triplet: dst[row, kt*C + c] = src[(bn*12+t+kt-1)*stride + off + c] (0-pad)
__global__ void gather3_kernel(const __bf16* __restrict__ src, int srcStride, int srcOff,
                               int C, __bf16* __restrict__ dst){
    long i = blockIdx.x*(long)blockDim.x + threadIdx.x;
    long tot = ROWS*(long)(3*C); if (i >= tot) return;
    int cc = (int)(i % (3*C)); long row = i / (3*C);
    int kt = cc / C, cch = cc % C;
    int t = (int)(row % 12); long bn = row / 12;
    int t2 = t + kt - 1;
    __bf16 v = (__bf16)0.f;
    if (t2 >= 0 && t2 < 12) v = src[(bn*12 + t2)*(long)srcStride + srcOff + cch];
    dst[i] = v;
}

// fused = relu( [p1m*sig(p1g) ; p2m*sig(p2g)] + x_init )
__global__ void conv_fuse_kernel(const float* __restrict__ p1, const float* __restrict__ p2,
                                 const __bf16* __restrict__ xinit, __bf16* __restrict__ fused){
    long i = blockIdx.x*(long)blockDim.x + threadIdx.x; if (i >= ROWS*128) return;
    int c = (int)(i & 127); long row = i >> 7;
    float po;
    if (c < 64) po = p1[row*128 + c]        * sigf(p1[row*128 + 64 + c]);
    else        po = p2[row*128 + (c-64)]   * sigf(p2[row*128 + c]);
    fused[i] = (__bf16)fmaxf(po + (float)xinit[i], 0.f);
}

// per-sequence attention: T=12, 4 heads, dk=16
__global__ void attn_kernel(const __bf16* __restrict__ q, const __bf16* __restrict__ k,
                            const __bf16* __restrict__ v, __bf16* __restrict__ o){
    const int s = blockIdx.x;                 // 16384 sequences
    __shared__ float qs[768], ks[768], vs[768];
    const int tid = threadIdx.x;              // 64 threads
    for (int i = tid; i < 768; i += 64){
        long g = (long)s*768 + i;
        qs[i] = (float)q[g]; ks[i] = (float)k[g]; vs[i] = (float)v[g];
    }
    __syncthreads();
    if (tid < 48){
        const int l = tid % 12, hd = tid / 12, off = hd*16;
        float sc[12]; float mx = -1e30f;
        for (int ss = 0; ss < 12; ++ss){
            float a = 0.f;
            for (int d = 0; d < 16; ++d) a += qs[l*64+off+d]*ks[ss*64+off+d];
            a *= 0.25f; sc[ss] = a; mx = fmaxf(mx, a);
        }
        float sum = 0.f;
        for (int ss = 0; ss < 12; ++ss){ sc[ss] = __expf(sc[ss]-mx); sum += sc[ss]; }
        const float inv = 1.f/sum;
        for (int d = 0; d < 16; ++d){
            float a = 0.f;
            for (int ss = 0; ss < 12; ++ss) a += sc[ss]*vs[ss*64+off+d];
            o[(long)s*768 + l*64 + off + d] = (__bf16)(a*inv);
        }
    }
}

__device__ __forceinline__ float ln_core(float v, int d, float* sm,
                                         const float* g, const float* bt){
    sm[d] = v; __syncthreads();
    for (int s = 32; s > 0; s >>= 1){ if (d < s) sm[d] += sm[d+s]; __syncthreads(); }
    float mean = sm[0]*(1.f/64.f); __syncthreads();
    float dev = v - mean;
    sm[d] = dev*dev; __syncthreads();
    for (int s = 32; s > 0; s >>= 1){ if (d < s) sm[d] += sm[d+s]; __syncthreads(); }
    float var = sm[0]*(1.f/64.f); __syncthreads();
    return dev*rsqrtf(var + 1e-5f)*g[d] + bt[d];
}
__global__ void ln_res_kernel(const float* __restrict__ x, const float* __restrict__ r,
                              const float* __restrict__ g, const float* __restrict__ bt,
                              float* __restrict__ outF, __bf16* __restrict__ outB){
    __shared__ float sm[64];
    long row = blockIdx.x; int d = threadIdx.x;
    float y = ln_core(x[row*64+d] + r[row*64+d], d, sm, g, bt);
    outF[row*64+d] = y; outB[row*64+d] = (__bf16)y;
}
__global__ void ln2_t_kernel(const float* __restrict__ x, const float* __restrict__ r,
                             const float* __restrict__ g, const float* __restrict__ bt,
                             __bf16* __restrict__ y2t){
    __shared__ float sm[64];
    long row = blockIdx.x; int d = threadIdx.x;
    float y = ln_core(x[row*64+d] + r[row*64+d], d, sm, g, bt);
    int t = (int)(row % 12); long bn = row / 12;
    int n = (int)(bn & 1023); int b = (int)(bn >> 10);
    y2t[((long)b*768 + t*64 + d)*1024 + n] = (__bf16)y;
}

// ---------------------------------------------------------------------------
static inline void launch_gemm(hipStream_t st, const __bf16* A, long sA,
                               const __bf16* Bt, long sB, int M, int N, int K,
                               const float* bias, int act,
                               float* oF, long sF, __bf16* oB, long sOB, int nb){
    int tiles = (M/32)*(N/32);
    dim3 grid((tiles + 7)/8, nb);
    gemm_wmma_bf16<<<grid, 256, 0, st>>>(A, sA, Bt, sB, M, N, K, bias, act, oF, sF, oB, sOB);
}

extern "C" void kernel_launch(void* const* d_in, const int* in_sizes, int n_in,
                              void* d_out, int out_size, void* d_ws, size_t ws_size,
                              hipStream_t stream){
    const float* flow  = (const float*)d_in[0];
    const int*   dayi  = (const int*)  d_in[1];
    const int*   weeki = (const int*)  d_in[2];
    const float* sp    = (const float*)d_in[3];
    const float* de    = (const float*)d_in[4];
    const float* we    = (const float*)d_in[5];
    const float* gfsW  = (const float*)d_in[6];   const float* gfsB = (const float*)d_in[7];
    const float* gluW  = (const float*)d_in[8];   const float* gluB = (const float*)d_in[9];
    const float* gcW   = (const float*)d_in[10];
    const float* gcA1  = (const float*)d_in[11];  const float* gcA2 = (const float*)d_in[12];
    const float* initW = (const float*)d_in[13];  const float* initB= (const float*)d_in[14];
    const float* p1mW  = (const float*)d_in[15];  const float* p1mB = (const float*)d_in[16];
    const float* p1gW  = (const float*)d_in[17];  const float* p1gB = (const float*)d_in[18];
    const float* p2mW  = (const float*)d_in[19];  const float* p2mB = (const float*)d_in[20];
    const float* p2gW  = (const float*)d_in[21];  const float* p2gB = (const float*)d_in[22];
    const float* dsfcW = (const float*)d_in[23];  const float* dsfcB= (const float*)d_in[24];
    const float* wq = (const float*)d_in[25]; const float* bq = (const float*)d_in[26];
    const float* wk = (const float*)d_in[27]; const float* bk = (const float*)d_in[28];
    const float* wv = (const float*)d_in[29]; const float* bv = (const float*)d_in[30];
    const float* wo = (const float*)d_in[31]; const float* bo = (const float*)d_in[32];
    const float* f1W = (const float*)d_in[33]; const float* f1B = (const float*)d_in[34];
    const float* f2W = (const float*)d_in[35]; const float* f2B = (const float*)d_in[36];
    const float* ln1g = (const float*)d_in[37]; const float* ln1b = (const float*)d_in[38];
    const float* ln2g = (const float*)d_in[39]; const float* ln2b = (const float*)d_in[40];
    const float* clw  = (const float*)d_in[41]; const float* clb  = (const float*)d_in[42];

    char* ws = (char*)d_ws; size_t off = 0;
    auto alloc = [&](size_t bytes)->void*{ void* p = ws + off; off += (bytes + 255) & ~(size_t)255; return p; };

    __bf16* w_gfs = (__bf16*)alloc(4096*2);
    __bf16* w_glu = (__bf16*)alloc(8192*2);
    __bf16* w_gc  = (__bf16*)alloc(4096*2);
    __bf16* w_ini = (__bf16*)alloc(24576*2);
    __bf16* w_p1  = (__bf16*)alloc(24576*2);
    __bf16* w_p2  = (__bf16*)alloc(24576*2);
    __bf16* w_fc  = (__bf16*)alloc(8192*2);
    __bf16* w_q   = (__bf16*)alloc(4096*2);
    __bf16* w_k   = (__bf16*)alloc(4096*2);
    __bf16* w_v   = (__bf16*)alloc(4096*2);
    __bf16* w_o   = (__bf16*)alloc(4096*2);
    __bf16* w_f1  = (__bf16*)alloc(16384*2);
    __bf16* w_f2  = (__bf16*)alloc(16384*2);
    float*  b_p1  = (float*)alloc(128*4);
    float*  b_p2  = (float*)alloc(128*4);
    float*  mvec  = (float*)alloc(1024*4);
    __bf16* Acomb = (__bf16*)alloc(1024L*1024*2);
    __bf16* fuseM = (__bf16*)alloc(1024L*1024*2);
    __bf16* S1    = (__bf16*)alloc(ROWS*64*2);   // xin -> hs -> y_bf16
    __bf16* S2    = (__bf16*)alloc(ROWS*64*2);   // x1 -> xg -> q -> y1_bf16
    __bf16* S3    = (__bf16*)alloc(ROWS*64*2);   // k
    __bf16* S4    = (__bf16*)alloc(ROWS*64*2);   // v
    __bf16* SBT   = (__bf16*)alloc(ROWS*64*2);   // x0t -> y2t  (== 16*768*1024)
    __bf16* SI    = (__bf16*)alloc(ROWS*128*2);  // x_init -> attn_out
    __bf16* SG3   = (__bf16*)alloc(ROWS*192*2);  // gather buf -> fused
    float*  F1    = (float*)alloc(ROWS*128*4);   // glu_g -> p1mg -> oproj -> ff2o
    float*  F2    = (float*)alloc(ROWS*128*4);   // p2mg -> ff1a(bf16)
    float*  YF    = (float*)alloc(ROWS*64*4);    // y f32
    float*  Y1F   = (float*)alloc(ROWS*64*4);    // y1 f32

    #define NB(n) ((int)(((n) + 255) / 256))
    // ---- weight prep ----
    f2bf_kernel<<<NB(4096),256,0,stream>>>(gfsW, w_gfs, 4096);
    f2bf_kernel<<<NB(8192),256,0,stream>>>(gluW, w_glu, 8192);
    transpose64_kernel<<<16,256,0,stream>>>(gcW, w_gc);
    conv3_reorder_kernel<<<NB(128*192),256,0,stream>>>(initW, initW, 128, 128, w_ini);
    conv3_reorder_kernel<<<NB(128*192),256,0,stream>>>(p1mW, p1gW, 64, 128, w_p1);
    conv3_reorder_kernel<<<NB(128*192),256,0,stream>>>(p2mW, p2gW, 64, 128, w_p2);
    f2bf_kernel<<<NB(8192),256,0,stream>>>(dsfcW, w_fc, 8192);
    f2bf_kernel<<<NB(4096),256,0,stream>>>(wq, w_q, 4096);
    f2bf_kernel<<<NB(4096),256,0,stream>>>(wk, w_k, 4096);
    f2bf_kernel<<<NB(4096),256,0,stream>>>(wv, w_v, 4096);
    f2bf_kernel<<<NB(4096),256,0,stream>>>(wo, w_o, 4096);
    f2bf_kernel<<<NB(16384),256,0,stream>>>(f1W, w_f1, 16384);
    f2bf_kernel<<<NB(16384),256,0,stream>>>(f2W, w_f2, 16384);
    concat2_kernel<<<1,128,0,stream>>>(p1mB, p1gB, 64, 128, b_p1);
    concat2_kernel<<<1,128,0,stream>>>(p2mB, p2gB, 64, 128, b_p2);
    softmax_add_kernel<<<1024,256,0,stream>>>(gcA1, gcA2, Acomb);
    node_mean_kernel<<<4,256,0,stream>>>(flow, clw, clb, mvec);
    fuse_mat_kernel<<<NB(1024L*1024),256,0,stream>>>(mvec, fuseM);

    // ---- embedding + FC + GLU ----
    embed_kernel<<<NB(ROWS*64),256,0,stream>>>(flow, dayi, weeki, sp, de, we, S1);
    launch_gemm(stream, S1, 0, w_gfs, 0, (int)ROWS, 64, 64, gfsB, 0, nullptr,0, S2,0, 1);
    launch_gemm(stream, S2, 0, w_glu, 0, (int)ROWS, 128, 64, gluB, 0, F1,0, nullptr,0, 1);
    glu_combine_kernel<<<NB(ROWS*64),256,0,stream>>>(F1, SBT);

    // ---- graph mix: hs[b] = Acomb @ x0[b]  (per batch 1024x1024 x 1024x768) ----
    launch_gemm(stream, Acomb, 0, SBT, 768L*1024, 1024, 768, 1024,
                nullptr, 0, nullptr,0, S1, 1024L*768, 16);
    // gc: xg = relu(hs @ gc_weight)
    launch_gemm(stream, S1, 0, w_gc, 0, (int)ROWS, 64, 64, nullptr, 1, nullptr,0, S2,0, 1);

    // ---- temporal conv block (as K=192 GEMMs) ----
    gather3_kernel<<<NB(ROWS*192),256,0,stream>>>(S2, 64, 0, 64, SG3);
    launch_gemm(stream, SG3, 0, w_ini, 0, (int)ROWS, 128, 192, initB, 0, nullptr,0, SI,0, 1);
    gather3_kernel<<<NB(ROWS*192),256,0,stream>>>(SI, 128, 0, 64, SG3);
    launch_gemm(stream, SG3, 0, w_p1, 0, (int)ROWS, 128, 192, b_p1, 0, F1,0, nullptr,0, 1);
    gather3_kernel<<<NB(ROWS*192),256,0,stream>>>(SI, 128, 64, 64, SG3);
    launch_gemm(stream, SG3, 0, w_p2, 0, (int)ROWS, 128, 192, b_p2, 0, F2,0, nullptr,0, 1);
    conv_fuse_kernel<<<NB(ROWS*128),256,0,stream>>>(F1, F2, SI, SG3);
    launch_gemm(stream, SG3, 0, w_fc, 0, (int)ROWS, 64, 128, dsfcB, 0, YF,0, S1,0, 1);

    // ---- attention ----
    launch_gemm(stream, S1, 0, w_q, 0, (int)ROWS, 64, 64, bq, 0, nullptr,0, S2,0, 1);
    launch_gemm(stream, S1, 0, w_k, 0, (int)ROWS, 64, 64, bk, 0, nullptr,0, S3,0, 1);
    launch_gemm(stream, S1, 0, w_v, 0, (int)ROWS, 64, 64, bv, 0, nullptr,0, S4,0, 1);
    attn_kernel<<<16384,64,0,stream>>>(S2, S3, S4, SI);
    launch_gemm(stream, SI, 0, w_o, 0, (int)ROWS, 64, 64, bo, 0, F1,0, nullptr,0, 1);
    ln_res_kernel<<<(int)ROWS,64,0,stream>>>(YF, F1, ln1g, ln1b, Y1F, S2);

    // ---- feed-forward + LN2 (write transposed y2t) ----
    launch_gemm(stream, S2, 0, w_f1, 0, (int)ROWS, 256, 64, f1B, 1, nullptr,0, (__bf16*)F2,0, 1);
    launch_gemm(stream, (__bf16*)F2, 0, w_f2, 0, (int)ROWS, 64, 256, f2B, 0, F1,0, nullptr,0, 1);
    ln2_t_kernel<<<(int)ROWS,64,0,stream>>>(Y1F, F1, ln2g, ln2b, SBT);

    // ---- final fuse mix: out[b] = fuse @ y2[b] ----
    launch_gemm(stream, fuseM, 0, SBT, 768L*1024, 1024, 768, 1024,
                nullptr, 0, (float*)d_out, 1024L*768, nullptr,0, 16);
    #undef NB
    (void)in_sizes; (void)n_in; (void)out_size; (void)ws_size;
}
